// ConformHopfieldBatchOneEnc_30794915512941
// MI455X (gfx1250) — compile-verified, matrix operand order checked by reading, not run on hardware
//
#include <hip/hip_runtime.h>
#include <hip/hip_bf16.h>

typedef __attribute__((ext_vector_type(16))) _Float16 v16h;
typedef __attribute__((ext_vector_type(8)))  float    v8f;

#define DEV __device__ __forceinline__

// ---------------- problem dimensions ----------------
constexpr int Bn = 16, Sn = 1024, DINn = 24, DOUTn = 128;
constexpr int Mh = 2, Hh = 4, CDn = 64, TOPK = 20, An = 6;
constexpr int HID = 200, HP = 224;           // hidden padded to 224 (7 x 32)
constexpr float SCALEf = 0.125f;             // 1/sqrt(64)
constexpr float NEGf = -1e30f;

constexpr float ALPH[An] = {0.05f, 0.06f, 0.08f, 0.10f, 0.12f, 0.14f};

// ---------------- output layout (flat, reference return order) ----------------
constexpr size_t OFF_SC  = 0;                                  // (2,)
constexpr size_t OFF_Y   = 2;                                  // (B,S,M)
constexpr size_t OFF_YP  = OFF_Y  + (size_t)Bn * Sn * Mh;      // (B,S,M)
constexpr size_t QOUT    = (size_t)Hh * Bn * An * Sn * Mh;     // 786432
constexpr size_t OFF_YLO = OFF_YP + (size_t)Bn * Sn * Mh;      // (H,B,A,S,M)
constexpr size_t OFF_YHI = OFF_YLO + QOUT;
constexpr size_t OFF_E   = OFF_YHI + QOUT;                     // (B,S,M)
constexpr size_t OFF_ELO = OFF_E + (size_t)Bn * Sn * Mh;
constexpr size_t OFF_EHI = OFF_ELO + QOUT;

// ---------------- WMMA helpers ----------------
union FragU { v16h v; uint4 q[2]; _Float16 h[16]; };

DEV v8f wmma_f16(v16h a, v16h b, v8f c) {
  // 8 args: (neg_a, A, neg_b, B, c_mod, C, reuse_a, reuse_b)
  return __builtin_amdgcn_wmma_f32_16x16x32_f16(false, a, false, b, (short)0, c,
                                                false, false);
}

// A-fragment (16x32, f16), row-major source with given element stride.
// lane l: row = l&15; two contiguous 8-elem chunks at K = 8*(l>>4) and 16+8*(l>>4).
// B-fragment uses the same pattern on pre-transposed weights W^T[N][K].
DEV v16h ld_frag(const _Float16* base, int stride) {
  const int l = threadIdx.x & 31;
  const _Float16* p = base + (size_t)(l & 15) * stride + ((l >> 4) << 3);
  FragU f;
  f.q[0] = *reinterpret_cast<const uint4*>(p);
  f.q[1] = *reinterpret_cast<const uint4*>(p + 16);
  return f.v;
}

// A-fragment from f32 X (rows x 24), K padded to 32 with zeros.
DEV v16h ld_fragA_x(const float* __restrict__ X, int rowBase) {
  const int l = threadIdx.x & 31, m = l & 15, half = l >> 4;
  const float* p = X + (size_t)(rowBase + m) * DINn;
  FragU f;
#pragma unroll
  for (int i = 0; i < 8; ++i) f.h[i] = (_Float16)p[8 * half + i];
  if (half == 0) {
#pragma unroll
    for (int i = 0; i < 8; ++i) f.h[8 + i] = (_Float16)p[16 + i];
  } else {
#pragma unroll
    for (int i = 0; i < 8; ++i) f.h[8 + i] = (_Float16)0.f;
  }
  return f.v;
}

DEV float quant20(const float* v, float t) {
  float p = t * 19.f;
  int i = (int)floorf(p);
  i = i < 0 ? 0 : (i > 18 ? 18 : i);
  float fr = p - (float)i;
  return v[i] + fr * (v[i + 1] - v[i]);
}

// ---------------- weight packing: dst[N][K] = src[K][N], zero padded ----------------
__global__ void pack_w_kernel(const float* __restrict__ src, _Float16* __restrict__ dst,
                              int Ksrc, int Nsrc, int Kpad, int Npad) {
  int i = blockIdx.x * 256 + threadIdx.x;
  if (i >= Kpad * Npad) return;
  int n = i / Kpad, k = i - n * Kpad;
  float v = (k < Ksrc && n < Nsrc) ? src[(size_t)k * Nsrc + n] : 0.f;
  dst[i] = (_Float16)v;
}

__global__ void pack_bias_kernel(const float* __restrict__ b1, const float* __restrict__ b2,
                                 const float* __restrict__ b3, float* __restrict__ b1p,
                                 float* __restrict__ b2p, float* __restrict__ b3p) {
  int i = blockIdx.x * 256 + threadIdx.x;
  if (i < HP)                    b1p[i] = (i < HID) ? b1[i] : 0.f;
  else if (i < 2 * HP)           { int j = i - HP;     b2p[j] = (j < HID) ? b2[j] : 0.f; }
  else if (i < 2 * HP + DOUTn)   { int j = i - 2 * HP; b3p[j] = b3[j]; }
}

// ---------------- encoder: 16 rows per wave through the full MLP ----------------
__global__ __launch_bounds__(32) void encoder_kernel(
    const float* __restrict__ Xtrue, const float* __restrict__ Xsim,
    const _Float16* __restrict__ W1t, const _Float16* __restrict__ W2t,
    const _Float16* __restrict__ W3t, const float* __restrict__ b1p,
    const float* __restrict__ b2p, const float* __restrict__ b3p,
    _Float16* __restrict__ encTrue, _Float16* __restrict__ encSim) {
  __shared__ __align__(16) _Float16 h1[16 * HP];
  __shared__ __align__(16) _Float16 h2[16 * HP];

  int tile = blockIdx.x;
  const float* X;
  _Float16* enc;
  if (tile < (Bn * Sn) / 16) { X = Xtrue; enc = encTrue; }
  else                       { X = Xsim;  enc = encSim; tile -= (Bn * Sn) / 16; }
  const int rowBase = tile * 16;
  const int l = threadIdx.x, m = l & 15, half = l >> 4;

  // zero the K-padding columns (200..223 -> only 208..223 need explicit zero;
  // cols 200..207 come out zero from zero-padded weights + relu)
#pragma unroll
  for (int i = 0; i < 8; ++i) {
    int idx = l * 8 + i;  // 0..255 covers 16 rows x 16 cols
    h1[(idx >> 4) * HP + 208 + (idx & 15)] = (_Float16)0.f;
    h2[(idx >> 4) * HP + 208 + (idx & 15)] = (_Float16)0.f;
  }

  // layer 1: (16x32) x (32x16) per output tile, 13 tiles cover 208 cols
  v16h ax = ld_fragA_x(X, rowBase);
  for (int nt = 0; nt < 13; ++nt) {
    v8f c = {};
    c = wmma_f16(ax, ld_frag(W1t + nt * 16 * 32, 32), c);
    const float bv = b1p[nt * 16 + m];
#pragma unroll
    for (int r = 0; r < 8; ++r) {
      float v = fmaxf(c[r] + bv, 0.f);
      h1[(r + 8 * half) * HP + nt * 16 + m] = (_Float16)v;
    }
  }
  __syncthreads();

  // layer 2: K = 224 (7 k-tiles), 13 output tiles
  v16h af[7];
#pragma unroll
  for (int kt = 0; kt < 7; ++kt) af[kt] = ld_frag(h1 + kt * 32, HP);
  for (int nt = 0; nt < 13; ++nt) {
    v8f c = {};
#pragma unroll
    for (int kt = 0; kt < 7; ++kt)
      c = wmma_f16(af[kt], ld_frag(W2t + (size_t)(nt * 16) * HP + kt * 32, HP), c);
    const float bv = b2p[nt * 16 + m];
#pragma unroll
    for (int r = 0; r < 8; ++r) {
      float v = fmaxf(c[r] + bv, 0.f);
      h2[(r + 8 * half) * HP + nt * 16 + m] = (_Float16)v;
    }
  }
  __syncthreads();

  // layer 3: K = 224, 8 output tiles (128 cols)
  v16h ag[7];
#pragma unroll
  for (int kt = 0; kt < 7; ++kt) ag[kt] = ld_frag(h2 + kt * 32, HP);
  for (int nt = 0; nt < 8; ++nt) {
    v8f c = {};
#pragma unroll
    for (int kt = 0; kt < 7; ++kt)
      c = wmma_f16(ag[kt], ld_frag(W3t + (size_t)(nt * 16) * HP + kt * 32, HP), c);
    const float bv = b3p[nt * 16 + m];
#pragma unroll
    for (int r = 0; r < 8; ++r)
      enc[(size_t)(rowBase + r + 8 * half) * DOUTn + nt * 16 + m] = (_Float16)(c[r] + bv);
  }
}

// ---------------- q/k projection: slice(64) @ W(64x256) -> [b][h][s][64] ----------------
__global__ __launch_bounds__(32) void proj_kernel(const _Float16* __restrict__ enc,
                                                  const _Float16* __restrict__ Wt,
                                                  _Float16* __restrict__ outp, int hop) {
  const int rowBase = blockIdx.x * 16;  // row in B*S
  const int l = threadIdx.x, m = l & 15, half = l >> 4;
  v16h a0 = ld_frag(enc + (size_t)rowBase * DOUTn + hop * CDn + 0,  DOUTn);
  v16h a1 = ld_frag(enc + (size_t)rowBase * DOUTn + hop * CDn + 32, DOUTn);
  for (int nt = 0; nt < 16; ++nt) {
    v8f c = {};
    c = wmma_f16(a0, ld_frag(Wt + (size_t)(nt * 16) * CDn + 0,  CDn), c);
    c = wmma_f16(a1, ld_frag(Wt + (size_t)(nt * 16) * CDn + 32, CDn), c);
    const int hh = nt >> 2;
    const int dBase = (nt & 3) * 16;
#pragma unroll
    for (int r = 0; r < 8; ++r) {
      const int s = rowBase + r + 8 * half;
      const int bb = s >> 10, ss = s & 1023;
      outp[((size_t)((bb * Hh + hh) * Sn) + ss) * CDn + dBase + m] = (_Float16)c[r];
    }
  }
}

// ---------------- scores (WMMA) + top-20 + quantiles + pinball ----------------
__global__ __launch_bounds__(64) void attn_topk_kernel(
    const _Float16* __restrict__ qb, const _Float16* __restrict__ kb,
    const float* __restrict__ errors, const float* __restrict__ y,
    const float* __restrict__ ypred, float* __restrict__ out,
    float* __restrict__ partial, int hop) {
  constexpr int SSTR = 520;  // 520 & 63 == 8 -> no LDS bank clumping
  __shared__ float scLds[16 * SSTR];
  __shared__ float candV[16 * 80];
  __shared__ int   candI[16 * 80];
  __shared__ float pinred[16];

  const int blk = blockIdx.x;  // b*256 + h*64 + qt
  const int qt = blk & 63;
  const int h  = (blk >> 6) & 3;
  const int b  = blk >> 8;
  const int qBase = qt * 16;

  const int tid = threadIdx.x;
  const int wave = tid >> 5;
  const int l = tid & 31, m = l & 15, half = l >> 4;

  const _Float16* qh = qb + (size_t)((b * Hh + h) * Sn) * CDn;
  const _Float16* kh = kb + (size_t)((b * Hh + h) * Sn) * CDn;

  v16h a0 = ld_frag(qh + (size_t)qBase * CDn + 0,  CDn);
  v16h a1 = ld_frag(qh + (size_t)qBase * CDn + 32, CDn);

  float tv[TOPK]; int ti[TOPK];
#pragma unroll
  for (int j = 0; j < TOPK; ++j) { tv[j] = -3.4e38f; ti[j] = 0; }

  const int row = tid >> 2;   // 0..15
  const int part = tid & 3;   // 0..3

  for (int hf = 0; hf < 2; ++hf) {
    // ---- WMMA: 16 queries x 512 keys into LDS (both waves, interleaved k-tiles)
    for (int kt = hf * 32 + wave; kt < hf * 32 + 32; kt += 2) {
      v8f c = {};
      c = wmma_f16(a0, ld_frag(kh + (size_t)(kt * 16) * CDn + 0,  CDn), c);
      c = wmma_f16(a1, ld_frag(kh + (size_t)(kt * 16) * CDn + 32, CDn), c);
      const int kcol = kt * 16 + m;
      const int lcol = kcol - hf * 512;
#pragma unroll
      for (int r = 0; r < 8; ++r) {
        const int Mq = r + 8 * half;
        float v = c[r] * SCALEf;
        if (qBase + Mq == kcol) v = NEGf;  // mask diagonal
        scLds[Mq * SSTR + lcol] = v;
      }
    }
    __syncthreads();
    // ---- streaming top-20 per quarter-row (register bubble-insert, static idx)
    const int cb = part * 128;
    for (int c0 = 0; c0 < 128; ++c0) {
      const int lc = cb + c0;
      float v = scLds[row * SSTR + lc];
      if (v > tv[TOPK - 1]) {
        float cv = v; int ci = hf * 512 + lc;
#pragma unroll
        for (int j = 0; j < TOPK; ++j) {
          if (cv > tv[j]) {
            float t0 = tv[j]; tv[j] = cv; cv = t0;
            int   t1 = ti[j]; ti[j] = ci; ci = t1;
          }
        }
      }
    }
    __syncthreads();
  }

#pragma unroll
  for (int j = 0; j < TOPK; ++j) {
    candV[row * 80 + part * 20 + j] = tv[j];
    candI[row * 80 + part * 20 + j] = ti[j];
  }
  __syncthreads();

  if (tid < 16) {
    const int rw = tid;
    int ptr[4] = {0, 0, 0, 0};
    float errv[TOPK];
    for (int t = 0; t < TOPK; ++t) {  // 4-way merge of sorted lists
      float bv = -3.5e38f; int bp = 0;
#pragma unroll
      for (int p = 0; p < 4; ++p) {
        if (ptr[p] < TOPK) {
          float vv = candV[rw * 80 + p * 20 + ptr[p]];
          if (vv > bv) { bv = vv; bp = p; }
        }
      }
      int idx = candI[rw * 80 + bp * 20 + ptr[bp]];
      ptr[bp]++;
      errv[t] = errors[(size_t)(b * Sn + idx) * 3 + hop];
    }
    // sort 20 errors ascending (static-index bubble)
#pragma unroll
    for (int i = 0; i < TOPK - 1; ++i)
#pragma unroll
      for (int j = 0; j < TOPK - 1; ++j)
        if (errv[j] > errv[j + 1]) { float t0 = errv[j]; errv[j] = errv[j + 1]; errv[j + 1] = t0; }

    const int qg = qBase + rw;
    const float yp = ypred[(size_t)(b * Sn + qg) * 3 + hop];
    const float yt = y[(size_t)(b * Sn + qg) * 3 + hop];
    float pinsum = 0.f;
#pragma unroll
    for (int a = 0; a < An; ++a) {
      const float tlo = ALPH[a] * 0.5f;
      const float thi = 1.0f - ALPH[a] * 0.5f;
      float qlo = quant20(errv, tlo);
      float qhi = quant20(errv, thi);
      float ylo = qlo + yp, yhi = qhi + yp;
      float dlo = yt - ylo, dhi = yt - yhi;
      pinsum += fmaxf(tlo * dlo, (tlo - 1.f) * dlo) + fmaxf(thi * dhi, (thi - 1.f) * dhi);
      const size_t o = ((((size_t)h * Bn + b) * An + a) * Sn + qg) * Mh + hop;
      out[OFF_YLO + o] = ylo;
      out[OFF_YHI + o] = yhi;
      out[OFF_ELO + o] = qlo;
      out[OFF_EHI + o] = qhi;
    }
    pinred[rw] = pinsum;
  }
  __syncthreads();
  if (tid == 0) {
    float s = 0.f;
    for (int i = 0; i < 16; ++i) s += pinred[i];
    partial[(size_t)hop * 4096 + blk] = s;  // deterministic per-block partial
  }
}

// ---------------- pass-through slices ----------------
__global__ void copy_slices_kernel(const float* __restrict__ y, const float* __restrict__ yp,
                                   const float* __restrict__ er, float* __restrict__ out) {
  int i = blockIdx.x * 256 + threadIdx.x;
  if (i >= 3 * Bn * Sn * Mh) return;
  int which = i / (Bn * Sn * Mh);
  int r = i - which * (Bn * Sn * Mh);
  int bs = r >> 1, hop = r & 1;
  const float* src = (which == 0) ? y : (which == 1) ? yp : er;
  size_t off = (which == 0) ? OFF_Y : (which == 1) ? OFF_YP : OFF_E;
  out[off + r] = src[(size_t)bs * 3 + hop];
}

// ---------------- deterministic final reduction of pinball means ----------------
__global__ __launch_bounds__(64) void reduce_scores_kernel(const float* __restrict__ partial,
                                                           float* __restrict__ out) {
  __shared__ float s[64];
  const int hop = blockIdx.x;
  const int NPART = Bn * Hh * (Sn / 16);  // 4096
  float acc = 0.f;
  for (int i = threadIdx.x; i < NPART; i += 64) acc += partial[(size_t)hop * NPART + i];
  s[threadIdx.x] = acc;
  __syncthreads();
  if (threadIdx.x == 0) {
    float t = 0.f;
    for (int i = 0; i < 64; ++i) t += s[i];
    out[OFF_SC + hop] = t / (float)((size_t)An * Bn * Hh * Sn);
  }
}

// ---------------- launcher ----------------
extern "C" void kernel_launch(void* const* d_in, const int* in_sizes, int n_in,
                              void* d_out, int out_size, void* d_ws, size_t ws_size,
                              hipStream_t stream) {
  (void)in_sizes; (void)n_in; (void)out_size; (void)ws_size;
  const float* Xtrue  = (const float*)d_in[0];
  const float* Xsim   = (const float*)d_in[1];
  const float* errors = (const float*)d_in[2];
  const float* y      = (const float*)d_in[3];
  const float* ypred  = (const float*)d_in[4];
  const float* W1 = (const float*)d_in[5];
  const float* b1 = (const float*)d_in[6];
  const float* W2 = (const float*)d_in[7];
  const float* b2 = (const float*)d_in[8];
  const float* W3 = (const float*)d_in[9];
  const float* b3 = (const float*)d_in[10];
  const float* Wq = (const float*)d_in[11];
  const float* Wk = (const float*)d_in[12];
  float* out = (float*)d_out;

  char* w = (char*)d_ws;
  size_t off = 0;
  auto carve = [&](size_t bytes) -> void* {
    void* p = w + off;
    off = (off + bytes + 255) & ~(size_t)255;
    return p;
  };
  _Float16* W1t = (_Float16*)carve((size_t)32 * HP * 2);
  _Float16* W2t = (_Float16*)carve((size_t)HP * HP * 2);
  _Float16* W3t = (_Float16*)carve((size_t)HP * DOUTn * 2);
  _Float16* Wqt = (_Float16*)carve((size_t)Mh * (Hh * CDn) * CDn * 2);
  _Float16* Wkt = (_Float16*)carve((size_t)Mh * (Hh * CDn) * CDn * 2);
  float* b1p = (float*)carve(HP * 4);
  float* b2p = (float*)carve(HP * 4);
  float* b3p = (float*)carve(DOUTn * 4);
  _Float16* encTrue = (_Float16*)carve((size_t)Bn * Sn * DOUTn * 2);
  _Float16* encSim  = (_Float16*)carve((size_t)Bn * Sn * DOUTn * 2);
  _Float16* qbuf = (_Float16*)carve((size_t)Bn * Hh * Sn * CDn * 2);
  _Float16* kbuf = (_Float16*)carve((size_t)Bn * Hh * Sn * CDn * 2);
  float* partial = (float*)carve((size_t)Mh * 4096 * 4);

  // pack weights (transposed, zero-padded) and biases
  pack_w_kernel<<<(32 * HP + 255) / 256, 256, 0, stream>>>(W1, W1t, DINn, HID, 32, HP);
  pack_w_kernel<<<(HP * HP + 255) / 256, 256, 0, stream>>>(W2, W2t, HID, HID, HP, HP);
  pack_w_kernel<<<(HP * DOUTn + 255) / 256, 256, 0, stream>>>(W3, W3t, HID, DOUTn, HP, DOUTn);
  for (int hop = 0; hop < Mh; ++hop) {
    pack_w_kernel<<<(CDn * Hh * CDn + 255) / 256, 256, 0, stream>>>(
        Wq + (size_t)hop * CDn * (Hh * CDn), Wqt + (size_t)hop * (Hh * CDn) * CDn,
        CDn, Hh * CDn, CDn, Hh * CDn);
    pack_w_kernel<<<(CDn * Hh * CDn + 255) / 256, 256, 0, stream>>>(
        Wk + (size_t)hop * CDn * (Hh * CDn), Wkt + (size_t)hop * (Hh * CDn) * CDn,
        CDn, Hh * CDn, CDn, Hh * CDn);
  }
  pack_bias_kernel<<<3, 256, 0, stream>>>(b1, b2, b3, b1p, b2p, b3p);

  // encoder over true+sim (2 * 16384 rows, 16 rows per wave)
  encoder_kernel<<<2 * (Bn * Sn) / 16, 32, 0, stream>>>(
      Xtrue, Xsim, W1t, W2t, W3t, b1p, b2p, b3p, encTrue, encSim);

  // per-hop: q/k projections then scores + top-K + quantiles
  for (int hop = 0; hop < Mh; ++hop) {
    proj_kernel<<<(Bn * Sn) / 16, 32, 0, stream>>>(
        encSim, Wqt + (size_t)hop * (Hh * CDn) * CDn, qbuf, hop);
    proj_kernel<<<(Bn * Sn) / 16, 32, 0, stream>>>(
        encTrue, Wkt + (size_t)hop * (Hh * CDn) * CDn, kbuf, hop);
    attn_topk_kernel<<<Bn * Hh * (Sn / 16), 64, 0, stream>>>(
        qbuf, kbuf, errors, y, ypred, out, partial, hop);
  }

  copy_slices_kernel<<<(3 * Bn * Sn * Mh + 255) / 256, 256, 0, stream>>>(y, ypred, errors, out);
  reduce_scores_kernel<<<Mh, 64, 0, stream>>>(partial, out);
}
